// FastSpeech_18820546691104
// MI455X (gfx1250) — matrix operand order repaired and checked by reading, not compiled, not run on hardware
//
#include <hip/hip_runtime.h>
#include <hip/hip_bf16.h>
#include <math.h>

// Problem constants (match reference)
#define BB 32
#define SS 256
#define TM 1024
#define HH 384
#define HD 192
#define NM 80
#define KW 5

typedef __attribute__((ext_vector_type(16))) _Float16 h16;
typedef __attribute__((ext_vector_type(8)))  float    f8;

__device__ __forceinline__ f8 wmma16(h16 a, h16 b, f8 c) {
  // D(16x16,f32) = A(16x32,f16) * B(32x16,f16) + C
  return __builtin_amdgcn_wmma_f32_16x16x32_f16(false, a, false, b, (short)0, c,
                                                false, false);
}

// ---- f16 fragment loaders (CDNA5 ISA 7.12.2 layouts) ----

// A: 16x32 tile from row-major f16, row stride ld (halves).
// lane m = lane&15; K chunk = 0..7 & 16..23 (lanes<16) or 8..15 & 24..31.
__device__ __forceinline__ h16 a_frag16(const _Float16* src, int ld, int lane) {
  int m  = lane & 15;
  int kb = (lane >> 4) << 3;
  const _Float16* p = src + (size_t)m * ld + kb;
  h16 a;
#pragma unroll
  for (int e = 0; e < 8; ++e) { a[e] = p[e]; a[e + 8] = p[e + 16]; }
  return a;
}

// B: 32x16 tile where storage is (N x K) row-major f16 (i.e. B = W^T).
// lane n = lane&15; K chunk = 0..7 & 16..23 (lanes<16) or 8..15 & 24..31.
__device__ __forceinline__ h16 b_frag_nk16(const _Float16* src, int ld, int lane) {
  int n  = lane & 15;
  int ks = (lane >> 4) << 3;
  const _Float16* p = src + (size_t)n * ld + ks;
  h16 b;
#pragma unroll
  for (int j = 0; j < 8; ++j) { b[j] = p[j]; b[j + 8] = p[j + 16]; }
  return b;
}

// ---- Weight prep: f32 -> f16 (straight) and (T,K,N) -> (T,N,K) transpose ----
__global__ void cvt16(const float* __restrict__ in, _Float16* __restrict__ out, int n) {
  int i = blockIdx.x * 256 + threadIdx.x;
  if (i < n) out[i] = (_Float16)in[i];
}

__global__ void transpose16(const float* __restrict__ in, _Float16* __restrict__ out,
                            int T, int K, int N) {
  int i = blockIdx.x * 256 + threadIdx.x;
  int tot = T * K * N;
  if (i >= tot) return;
  int t = i / (K * N);
  int r = i % (K * N);
  int k = r / N;
  int n = r % N;
  out[(size_t)t * N * K + (size_t)n * K + k] = (_Float16)in[i];
}

// ---- V transpose: qkv16 (B,S,3H) -> vT (B, HEADS, HD, S) so att@V B-fragments
//      become two contiguous 8-half runs per lane (no strided 2-byte gathers). ----
__global__ void vtrans_kernel(const _Float16* __restrict__ qkv, _Float16* __restrict__ vT,
                              int Scur) {
  int i = blockIdx.x * 256 + threadIdx.x;
  int tot = BB * Scur * HH;
  if (i >= tot) return;
  int d0 = i % HH;
  int bs = i / HH;
  int b = bs / Scur, s = bs % Scur;
  int hh = d0 / HD, d = d0 % HD;
  vT[((size_t)(b * 2 + hh) * HD + d) * Scur + s] =
      qkv[(size_t)bs * (3 * HH) + 2 * HH + d0];
}

// ---- GEMM: Y(MxN) = X16(MxK) @ W16^T (+bias); W16 stored (N x K).
//      Pure f16 loads -> wmma; emits f32 and/or f16 outputs. ----
__global__ void gemm16(const _Float16* __restrict__ X, const _Float16* __restrict__ W,
                       const float* __restrict__ bias, float* __restrict__ Y32,
                       _Float16* __restrict__ Y16, int M, int N, int K) {
  int lane = threadIdx.x & 31;
  int gw   = (blockIdx.x << 2) + (threadIdx.x >> 5);
  int nt   = N >> 4;
  int mt   = M >> 4;
  if (gw >= mt * nt) return;           // wave-uniform; EXEC all-ones inside
  int m0 = (gw / nt) << 4;
  int n0 = (gw % nt) << 4;
  f8 c;
#pragma unroll
  for (int i = 0; i < 8; ++i) c[i] = 0.f;
  for (int k0 = 0; k0 < K; k0 += 32) {
    h16 a = a_frag16(X + (size_t)m0 * K + k0, K, lane);
    h16 b = b_frag_nk16(W + (size_t)n0 * K + k0, K, lane);
    c = wmma16(a, b, c);
  }
  int n = lane & 15, hi = lane >> 4;
  float bv = bias[n0 + n];
#pragma unroll
  for (int v = 0; v < 8; ++v) {
    float y = c[v] + bv;
    size_t idx = (size_t)(m0 + v + (hi << 3)) * N + n0 + n;
    if (Y32) Y32[idx] = y;
    if (Y16) Y16[idx] = (_Float16)y;
  }
}

// ---- Conv1d 'same', K=5, weights pre-transposed to (5, N, K) f16 ----
__global__ void conv1d16(const _Float16* __restrict__ X, const _Float16* __restrict__ Wt,
                         const float* __restrict__ bias, float* __restrict__ Y,
                         int Scur) {
  int lane = threadIdx.x & 31;
  int gw   = (blockIdx.x << 2) + (threadIdx.x >> 5);
  const int nt = HH >> 4;  // 24
  int mt = (BB * Scur) >> 4;
  if (gw >= mt * nt) return;
  int m0 = (gw / nt) << 4;
  int n0 = (gw % nt) << 4;
  int b = m0 / Scur, s0 = m0 % Scur;   // tiles never cross batch (Scur % 16 == 0)
  const _Float16* xb = X + (size_t)b * Scur * HH;
  f8 c;
#pragma unroll
  for (int i = 0; i < 8; ++i) c[i] = 0.f;
  int m = lane & 15, kb = (lane >> 4) << 3;
  for (int t = 0; t < KW; ++t) {
    int s = s0 + m + (t - 2);
    bool in = (s >= 0) && (s < Scur);
    const _Float16* prow = xb + (size_t)(in ? s : 0) * HH;
    const _Float16* wt = Wt + (size_t)t * HH * HH;  // (N,K) layout
    for (int k0 = 0; k0 < HH; k0 += 32) {
      h16 a;
      if (in) {
        const _Float16* p = prow + k0 + kb;
#pragma unroll
        for (int e = 0; e < 8; ++e) { a[e] = p[e]; a[e + 8] = p[e + 16]; }
      } else {
#pragma unroll
        for (int e = 0; e < 16; ++e) a[e] = (_Float16)0.f;
      }
      h16 bf = b_frag_nk16(wt + (size_t)n0 * HH + k0, HH, lane);
      c = wmma16(a, bf, c);
    }
  }
  int n = lane & 15, hi = lane >> 4;
  float bv = bias[n0 + n];
#pragma unroll
  for (int v = 0; v < 8; ++v)
    Y[(size_t)(m0 + v + (hi << 3)) * HH + n0 + n] = c[v] + bv;
}

// ---- Fused 2-head attention; q/k from f16 qkv, V from transposed copy.
//      One wave per 16-row query tile per (b,h). All loads contiguous. ----
__global__ void attention16(const _Float16* __restrict__ qkv,
                            const _Float16* __restrict__ vT,
                            _Float16* __restrict__ out, int Scur) {
  const int ld = 3 * HH;
  int lane = threadIdx.x & 31;
  int wv   = threadIdx.x >> 5;
  int gw   = (blockIdx.x << 2) + wv;
  int mtiles = Scur >> 4;
  if (gw >= BB * 2 * mtiles) return;
  int bh = gw / mtiles, mt = gw % mtiles;
  int b = bh >> 1, h = bh & 1;
  const _Float16* qp  = qkv + (size_t)(b * Scur + mt * 16) * ld + h * HD;
  const _Float16* kp  = qkv + (size_t)b * Scur * ld + HH + h * HD;
  const _Float16* vtp = vT + (size_t)(b * 2 + h) * HD * Scur;  // (HD, S) row-major
  const float scale = 0.07216878364870323f;  // 1/sqrt(192)

  int m = lane & 15, kb8 = (lane >> 4) << 3, ks16 = (lane >> 4) << 4;
  int n = lane & 15, hi = lane >> 4;

  // Pre-scaled q fragments in registers: 6 chunks of K=32 over hd=192.
  h16 aq[6];
#pragma unroll
  for (int ch = 0; ch < 6; ++ch) {
    const _Float16* p = qp + (size_t)m * ld + ch * 32 + kb8;
#pragma unroll
    for (int e = 0; e < 8; ++e) {
      aq[ch][e]     = (_Float16)((float)p[e] * scale);
      aq[ch][e + 8] = (_Float16)((float)p[e + 16] * scale);
    }
  }

  // Pass 1: row max over all keys
  float vmax[8];
#pragma unroll
  for (int v = 0; v < 8; ++v) vmax[v] = -1e30f;
  for (int kt = 0; kt < mtiles; ++kt) {
    f8 sc;
#pragma unroll
    for (int i = 0; i < 8; ++i) sc[i] = 0.f;
    const _Float16* kt_p = kp + (size_t)(kt * 16) * ld;
#pragma unroll
    for (int ch = 0; ch < 6; ++ch) {
      h16 bk = b_frag_nk16(kt_p + ch * 32, ld, lane);  // B[d][n] = K[key=n][d]
      sc = wmma16(aq[ch], bk, sc);
    }
#pragma unroll
    for (int v = 0; v < 8; ++v) vmax[v] = fmaxf(vmax[v], sc[v]);
  }
#pragma unroll
  for (int off = 8; off > 0; off >>= 1)
#pragma unroll
    for (int v = 0; v < 8; ++v) vmax[v] = fmaxf(vmax[v], __shfl_xor(vmax[v], off, 32));

  // Pass 2: exp + rowsum + att@V (pairs of 16-key tiles -> one K=32 A fragment)
  float vsum[8];
#pragma unroll
  for (int v = 0; v < 8; ++v) vsum[v] = 0.f;
  f8 oacc[12];
#pragma unroll
  for (int t2 = 0; t2 < 12; ++t2)
#pragma unroll
    for (int i = 0; i < 8; ++i) oacc[t2][i] = 0.f;

  __shared__ _Float16 ab[4][16][34];
  _Float16 (*abw)[34] = ab[wv];

  int kpairs = Scur >> 5;
  for (int kt2 = 0; kt2 < kpairs; ++kt2) {
    for (int half = 0; half < 2; ++half) {
      int kt = kt2 * 2 + half;
      f8 sc;
#pragma unroll
      for (int i = 0; i < 8; ++i) sc[i] = 0.f;
      const _Float16* kt_p = kp + (size_t)(kt * 16) * ld;
#pragma unroll
      for (int ch = 0; ch < 6; ++ch) {
        h16 bk = b_frag_nk16(kt_p + ch * 32, ld, lane);
        sc = wmma16(aq[ch], bk, sc);
      }
#pragma unroll
      for (int v = 0; v < 8; ++v) {
        float ev = __expf(sc[v] - vmax[v]);
        vsum[v] += ev;
        abw[v + (hi << 3)][(half << 4) + n] = (_Float16)ev;  // wave-local LDS; DS in-order
      }
    }
    h16 aa;
#pragma unroll
    for (int e = 0; e < 8; ++e) {
      aa[e]     = abw[m][kb8 + e];
      aa[e + 8] = abw[m][kb8 + 16 + e];
    }
    // att@V: B[kd][n] = V[key][dim] = vT[dim][key]; vT read is contiguous in key.
#pragma unroll
    for (int nt2 = 0; nt2 < 12; ++nt2) {
      h16 bv;
      const _Float16* p = vtp + (size_t)(nt2 * 16 + n) * Scur + kt2 * 32 + kb8;
#pragma unroll
      for (int j = 0; j < 8; ++j) { bv[j] = p[j]; bv[j + 8] = p[j + 16]; }
      oacc[nt2] = wmma16(aa, bv, oacc[nt2]);
    }
  }
#pragma unroll
  for (int off = 8; off > 0; off >>= 1)
#pragma unroll
    for (int v = 0; v < 8; ++v) vsum[v] += __shfl_xor(vsum[v], off, 32);

#pragma unroll
  for (int nt2 = 0; nt2 < 12; ++nt2)
#pragma unroll
    for (int v = 0; v < 8; ++v)
      out[(size_t)(b * Scur + mt * 16 + v + (hi << 3)) * HH + h * HD + nt2 * 16 + n] =
          (_Float16)(oacc[nt2][v] / vsum[v]);
}

// ---- out = LN(t (+ r)) * g + b, optional relu; writes f32 + optional f16 mirror ----
__global__ void add_ln(const float* __restrict__ t, const float* __restrict__ r,
                       const float* __restrict__ g, const float* __restrict__ be,
                       float* __restrict__ out, _Float16* __restrict__ out16, int relu) {
  int row = blockIdx.x;
  int tid = threadIdx.x;
  const float* tp = t + (size_t)row * HH;
  const float* rp = r ? r + (size_t)row * HH : nullptr;
  float v[3];
  float s = 0.f;
#pragma unroll
  for (int i = 0; i < 3; ++i) {
    int idx = tid + i * 128;
    float x = tp[idx] + (rp ? rp[idx] : 0.f);
    v[i] = x; s += x;
  }
  __shared__ float red[128];
  red[tid] = s; __syncthreads();
  for (int o = 64; o > 0; o >>= 1) { if (tid < o) red[tid] += red[tid + o]; __syncthreads(); }
  float mu = red[0] * (1.0f / HH);
  __syncthreads();
  float s2 = 0.f;
#pragma unroll
  for (int i = 0; i < 3; ++i) { float d = v[i] - mu; s2 += d * d; }
  red[tid] = s2; __syncthreads();
  for (int o = 64; o > 0; o >>= 1) { if (tid < o) red[tid] += red[tid + o]; __syncthreads(); }
  float inv = rsqrtf(red[0] * (1.0f / HH) + 1e-5f);
#pragma unroll
  for (int i = 0; i < 3; ++i) {
    int idx = tid + i * 128;
    float y = (v[i] - mu) * inv * g[idx] + be[idx];
    if (relu) y = fmaxf(y, 0.f);
    out[(size_t)row * HH + idx] = y;
    if (out16) out16[(size_t)row * HH + idx] = (_Float16)y;
  }
}

// Positional encoding value (quirk: pos = batch index in the reference)
__device__ __forceinline__ float pe_val(int pos, int hcol) {
  const float cc = 9.210340371976184f / 384.0f;  // ln(10000)/H
  float den = __expf(-(float)(hcol & ~1) * cc);
  float a = (float)pos * den;
  return (hcol & 1) ? __cosf(a) : __sinf(a);
}

__global__ void embed_kernel(const int* __restrict__ tok, const float* __restrict__ emb,
                             float* __restrict__ x, _Float16* __restrict__ x16) {
  int bs = blockIdx.x;
  int b = bs / SS;
  int t = tok[bs];
  const float* er = emb + (size_t)t * HH;
#pragma unroll
  for (int i = 0; i < 3; ++i) {
    int h = threadIdx.x + i * 128;
    float v = er[h] + pe_val(b, h);
    x[(size_t)bs * HH + h] = v;
    x16[(size_t)bs * HH + h] = (_Float16)v;
  }
}

__global__ void cumsum_kernel(const float* __restrict__ al, const int* __restrict__ gt,
                              float* __restrict__ cum) {
  int b = blockIdx.x;
  if (threadIdx.x == 0) {
    float c = 0.f, g = (float)gt[b];
    for (int s = 0; s < SS; ++s) { c += al[b * SS + s]; cum[b * SS + s] = fminf(c, g); }
  }
}

__global__ void expand_kernel(const float* __restrict__ x, const float* __restrict__ cum,
                              float* __restrict__ y, _Float16* __restrict__ y16) {
  int bm = blockIdx.x;
  int b = bm / TM, mfr = bm % TM;
  const float* cr = cum + (size_t)b * SS;
  float fm = (float)mfr;
  int lo = 0, hiq = SS;
  while (lo < hiq) { int mid = (lo + hiq) >> 1; if (cr[mid] <= fm) lo = mid + 1; else hiq = mid; }
  bool valid = fm < cr[SS - 1];
  int idx = lo < SS ? lo : SS - 1;
  const float* xr = x + (size_t)(b * SS + idx) * HH;
#pragma unroll
  for (int i = 0; i < 3; ++i) {
    int h = threadIdx.x + i * 128;
    float v = (valid ? xr[h] : 0.f) + pe_val(b, h);
    y[(size_t)bm * HH + h] = v;
    y16[(size_t)bm * HH + h] = (_Float16)v;
  }
}

__global__ void duration_kernel(const float* __restrict__ m, const float* __restrict__ lw,
                                const float* __restrict__ lb, const int* __restrict__ tn,
                                float* __restrict__ align_pred, float* __restrict__ num_pred) {
  int b = blockIdx.x, s = threadIdx.x;
  const float* row = m + (size_t)(b * SS + s) * HH;
  float d = lb[0];
  for (int h = 0; h < HH; ++h) d += row[h] * lw[h];
  bool mk = s < tn[b];
  align_pred[b * SS + s] = mk ? d : 0.f;
  float e = mk ? __expf(d) : 0.f;
  __shared__ float red[256];
  red[s] = e; __syncthreads();
  for (int o = 128; o > 0; o >>= 1) { if (s < o) red[s] += red[s + o]; __syncthreads(); }
  if (s == 0) num_pred[b] = (float)(int)red[0];
}

// ---- Host-side transformer block driver ----
static void run_block(float* x32, _Float16* x16, _Float16* qkv16, _Float16* atto16,
                      _Float16* vT16, float* tmp32, int Scur,
                      const _Float16* in_w16, const float* in_b,
                      const _Float16* out_w16, const float* out_b,
                      const float* ln1g, const float* ln1b,
                      const _Float16* cw16t, const float* cb,
                      const float* ln2g, const float* ln2b, hipStream_t stream) {
  int M = BB * Scur;
  {
    int waves = (M / 16) * ((3 * HH) / 16);
    gemm16<<<(waves + 3) / 4, 128, 0, stream>>>(x16, in_w16, in_b, (float*)nullptr,
                                                qkv16, M, 3 * HH, HH);
  }
  {
    int tot = M * HH;
    vtrans_kernel<<<(tot + 255) / 256, 256, 0, stream>>>(qkv16, vT16, Scur);
  }
  {
    int waves = BB * 2 * (Scur / 16);
    attention16<<<(waves + 3) / 4, 128, 0, stream>>>(qkv16, vT16, atto16, Scur);
  }
  {
    int waves = (M / 16) * (HH / 16);
    gemm16<<<(waves + 3) / 4, 128, 0, stream>>>(atto16, out_w16, out_b, tmp32,
                                                (_Float16*)nullptr, M, HH, HH);
  }
  add_ln<<<M, 128, 0, stream>>>(tmp32, x32, ln1g, ln1b, x32, x16, 0);
  {
    int waves = (M / 16) * (HH / 16);
    conv1d16<<<(waves + 3) / 4, 128, 0, stream>>>(x16, cw16t, cb, tmp32, Scur);
  }
  add_ln<<<M, 128, 0, stream>>>(tmp32, x32, ln2g, ln2b, x32, x16, 0);
}

extern "C" void kernel_launch(void* const* d_in, const int* in_sizes, int n_in,
                              void* d_out, int out_size, void* d_ws, size_t ws_size,
                              hipStream_t stream) {
  (void)in_sizes; (void)n_in; (void)out_size; (void)ws_size;
  const int*   tokens     = (const int*)d_in[0];
  const int*   token_nums = (const int*)d_in[1];
  const float* align      = (const float*)d_in[2];
  const int*   gt         = (const int*)d_in[3];
  const float* emb        = (const float*)d_in[4];
  const float* e_in_w  = (const float*)d_in[5];
  const float* e_in_b  = (const float*)d_in[6];
  const float* e_out_w = (const float*)d_in[7];
  const float* e_out_b = (const float*)d_in[8];
  const float* e_ln1g  = (const float*)d_in[9];
  const float* e_ln1b  = (const float*)d_in[10];
  const float* e_cw    = (const float*)d_in[11];
  const float* e_cb    = (const float*)d_in[12];
  const float* e_ln2g  = (const float*)d_in[13];
  const float* e_ln2b  = (const float*)d_in[14];
  const float* dd_in_w  = (const float*)d_in[15];
  const float* dd_in_b  = (const float*)d_in[16];
  const float* dd_out_w = (const float*)d_in[17];
  const float* dd_out_b = (const float*)d_in[18];
  const float* dd_ln1g  = (const float*)d_in[19];
  const float* dd_ln1b  = (const float*)d_in[20];
  const float* dd_cw    = (const float*)d_in[21];
  const float* dd_cb    = (const float*)d_in[22];
  const float* dd_ln2g  = (const float*)d_in[23];
  const float* dd_ln2b  = (const float*)d_in[24];
  const float* du_cw  = (const float*)d_in[25];
  const float* du_cb  = (const float*)d_in[26];
  const float* du_lng = (const float*)d_in[27];
  const float* du_lnb = (const float*)d_in[28];
  const float* du_lw  = (const float*)d_in[29];
  const float* du_lb  = (const float*)d_in[30];
  const float* mel_w  = (const float*)d_in[31];
  const float* mel_b  = (const float*)d_in[32];

  // ---- Workspace layout: f32 region then f16 region (~277 MB total) ----
  float* f32b   = (float*)d_ws;
  float* x_enc32 = f32b;                               // B*S*H
  float* x_dec32 = x_enc32 + (size_t)BB * SS * HH;     // B*T*H
  float* tmp32   = x_dec32 + (size_t)BB * TM * HH;     // B*T*H
  float* cum     = tmp32 + (size_t)BB * TM * HH;       // B*S

  _Float16* h = (_Float16*)(cum + (size_t)BB * SS);
  _Float16* x_enc16 = h;                               // B*S*H
  _Float16* x_dec16 = x_enc16 + (size_t)BB * SS * HH;  // B*T*H
  _Float16* qkv16   = x_dec16 + (size_t)BB * TM * HH;  // B*T*3H
  _Float16* atto16  = qkv16 + (size_t)BB * TM * 3 * HH;// B*T*H
  _Float16* vT16    = atto16 + (size_t)BB * TM * HH;   // B*T*H (transposed V)
  _Float16* w_e_in  = vT16 + (size_t)BB * TM * HH;     // 3H*H
  _Float16* w_e_out = w_e_in + (size_t)3 * HH * HH;    // H*H
  _Float16* w_e_cv  = w_e_out + (size_t)HH * HH;       // 5*H*H (transposed)
  _Float16* w_d_in  = w_e_cv + (size_t)KW * HH * HH;
  _Float16* w_d_out = w_d_in + (size_t)3 * HH * HH;
  _Float16* w_d_cv  = w_d_out + (size_t)HH * HH;
  _Float16* w_u_cv  = w_d_cv + (size_t)KW * HH * HH;
  _Float16* w_mel   = w_u_cv + (size_t)KW * HH * HH;   // NM*H (transposed)

  float* mel_out = (float*)d_out;                      // B*T*NM
  float* ap_out  = mel_out + (size_t)BB * TM * NM;     // B*S
  float* np_out  = ap_out + (size_t)BB * SS;           // B

  // ---- Weight prep (f16 staging; transposes make all B fragments contiguous) ----
  {
    int n1 = 3 * HH * HH;  // 442368
    cvt16<<<(n1 + 255) / 256, 256, 0, stream>>>(e_in_w, w_e_in, n1);
    cvt16<<<(n1 + 255) / 256, 256, 0, stream>>>(dd_in_w, w_d_in, n1);
    int n2 = HH * HH;      // 147456
    cvt16<<<(n2 + 255) / 256, 256, 0, stream>>>(e_out_w, w_e_out, n2);
    cvt16<<<(n2 + 255) / 256, 256, 0, stream>>>(dd_out_w, w_d_out, n2);
    int n3 = KW * HH * HH; // 737280
    transpose16<<<(n3 + 255) / 256, 256, 0, stream>>>(e_cw, w_e_cv, KW, HH, HH);
    transpose16<<<(n3 + 255) / 256, 256, 0, stream>>>(dd_cw, w_d_cv, KW, HH, HH);
    transpose16<<<(n3 + 255) / 256, 256, 0, stream>>>(du_cw, w_u_cv, KW, HH, HH);
    int n4 = HH * NM;      // 30720
    transpose16<<<(n4 + 255) / 256, 256, 0, stream>>>(mel_w, w_mel, 1, HH, NM);
  }

  // ---- Encoder ----
  embed_kernel<<<BB * SS, 128, 0, stream>>>(tokens, emb, x_enc32, x_enc16);
  run_block(x_enc32, x_enc16, qkv16, atto16, vT16, tmp32, SS,
            w_e_in, e_in_b, w_e_out, e_out_b, e_ln1g, e_ln1b,
            w_e_cv, e_cb, e_ln2g, e_ln2b, stream);

  // ---- Duration predictor: conv -> LN+relu (in-place on tmp32) -> linear/mask ----
  {
    int M = BB * SS;
    int waves = (M / 16) * (HH / 16);
    conv1d16<<<(waves + 3) / 4, 128, 0, stream>>>(x_enc16, w_u_cv, du_cb, tmp32, SS);
    add_ln<<<M, 128, 0, stream>>>(tmp32, (const float*)nullptr, du_lng, du_lnb,
                                  tmp32, (_Float16*)nullptr, 1);
    duration_kernel<<<BB, 256, 0, stream>>>(tmp32, du_lw, du_lb, token_nums, ap_out, np_out);
  }

  // ---- Length regulator ----
  cumsum_kernel<<<BB, 32, 0, stream>>>(align, gt, cum);
  expand_kernel<<<BB * TM, 128, 0, stream>>>(x_enc32, cum, x_dec32, x_dec16);

  // ---- Decoder ----
  run_block(x_dec32, x_dec16, qkv16, atto16, vT16, tmp32, TM,
            w_d_in, dd_in_b, w_d_out, dd_out_b, dd_ln1g, dd_ln1b,
            w_d_cv, dd_cb, dd_ln2g, dd_ln2b, stream);

  // ---- Mel projection: (B*T, H) @ mel_w (+bias) -> d_out ----
  {
    int M = BB * TM;
    int waves = (M / 16) * (NM / 16);
    gemm16<<<(waves + 3) / 4, 128, 0, stream>>>(x_dec16, w_mel, mel_b, mel_out,
                                                (_Float16*)nullptr, M, NM, HH);
  }
}
